// BeamDecoder_56092272886281
// MI455X (gfx1250) — compile-verified, matrix operand order checked by reading, not run on hardware
//
#include <hip/hip_runtime.h>
#include <hip/hip_bf16.h>
#include <math.h>

#define VSZ     50257
#define ESZ     128
#define H1SZ    128
#define H2SZ    64
#define QSZ     64
#define KB      10
#define TENC    2048
#define MAXLEN  30
#define SOSI    1
#define EOSI    2
#define NEGF    (-1e30f)
#define VPAD    50272          // 3142 * 16
#define NTILES  3142
#define TPW     8              // N-tiles per wave in the WMMA kernel
#define NWBLK   ((NTILES + TPW - 1) / TPW)   // 393
#define MSLICE  64             // slices for partial max/sum
#define MCH     786            // ceil(VSZ / MSLICE)
#define NTB     40             // blocks for per-block top-10
#define TOTC    (KB * VSZ)     // 502570 candidates
#define NC      (NTB * KB)     // 400 merged candidates

typedef __attribute__((ext_vector_type(16))) _Float16 v16h;
typedef __attribute__((ext_vector_type(8)))  _Float16 v8h;
typedef __attribute__((ext_vector_type(8)))  float    v8f;

__device__ __forceinline__ float sigf(float x) { return 1.0f / (1.0f + __expf(-x)); }

// ---------------- workspace layout (bytes) ----------------
// wc16   : 0           VPAD*128 f16 = 12,869,632   (rows >= V zero-padded)
// x16    : 12,869,632  16*128 f16   = 4,096
// logits : 12,873,728  16*VPAD f32  = 3,217,408
// stF    : 16,091,136  4490 floats
// stI    : 16,109,096  330 ints
// partM  : 16,110,416  MSLICE*KB floats
// partS  : 16,112,976  MSLICE*KB floats
// topv   : 16,115,536  NC floats
// topi   : 16,117,136  NC ints
// bcp    : 16,118,736  VPAD floats (zero-padded bias)
#define OFF_X16    12869632
#define OFF_LOG    12873728
#define OFF_STF    16091136
#define OFF_STI    16109096
#define OFF_PM     16110416
#define OFF_PS     16112976
#define OFF_TV     16115536
#define OFF_TI     16117136
#define OFF_BC     16118736

// ---------------------------------------------------------------------------
// Convert Wc -> f16 padded to VPAD rows; also produce zero-padded bias copy.
__global__ __launch_bounds__(256) void k_cvt_wc(const float* __restrict__ Wc,
                                                const float* __restrict__ bc,
                                                _Float16* __restrict__ wc16,
                                                float* __restrict__ bcp) {
  int i = blockIdx.x * blockDim.x + threadIdx.x;
  if (i < VPAD * 128)
    wc16[i] = (i < VSZ * 128) ? (_Float16)Wc[i] : (_Float16)0.0f;
  if (i < VPAD)
    bcp[i] = (i < VSZ) ? bc[i] : 0.0f;
}

__global__ __launch_bounds__(256) void k_init_state(float* stF, int* stI) {
  int tid = threadIdx.x;
  for (int i = tid; i < 4480; i += 256) stF[i] = 0.0f;
  if (tid < KB) { stF[4480 + tid] = 0.0f; stI[tid] = SOSI; stI[KB + tid] = 0; }
  for (int i = tid; i < KB * (MAXLEN + 1); i += 256) stI[2 * KB + i] = SOSI;
}

// ---------------------------------------------------------------------------
// One step of the recurrent decoder (LSTM1, LSTM2, attention) for all beams.
__global__ __launch_bounds__(256) void k_step_rnn(
    const float* __restrict__ emb,
    const float* __restrict__ Wih1, const float* __restrict__ Whh1,
    const float* __restrict__ bih1, const float* __restrict__ bhh1,
    const float* __restrict__ Wih2, const float* __restrict__ Whh2,
    const float* __restrict__ bih2, const float* __restrict__ bhh2,
    const float* __restrict__ Wq,  const float* __restrict__ bq,
    const float* __restrict__ enc_key, const float* __restrict__ enc_val,
    const float* __restrict__ mask,
    float* __restrict__ stF, const int* __restrict__ stI,
    _Float16* __restrict__ x16) {
  const int tid = threadIdx.x;
  float* h1 = stF;            float* c1 = stF + 1280;
  float* h2 = stF + 2560;     float* c2 = stF + 3200;
  float* ctx = stF + 3840;
  const int* prev = stI;

  __shared__ float sx[KB][ESZ + QSZ];
  __shared__ float sh1[KB][H1SZ], sc1[KB][H1SZ];
  __shared__ float sh2[KB][H2SZ], sc2[KB][H2SZ];
  __shared__ float sq[KB][QSZ];
  __shared__ float sctx[KB][QSZ];
  __shared__ float se[TENC];
  __shared__ float sred[256];
  __shared__ float spart[4][QSZ];

  for (int i = tid; i < KB * ESZ; i += 256) {
    int b = i / ESZ, j = i % ESZ;
    sx[b][j] = emb[(size_t)prev[b] * ESZ + j];
  }
  for (int i = tid; i < KB * QSZ; i += 256) {
    int b = i / QSZ, j = i % QSZ;
    float cv = ctx[i];
    sx[b][ESZ + j] = cv;
    (&sctx[0][0])[i] = cv;
  }
  for (int i = tid; i < KB * H1SZ; i += 256) { (&sh1[0][0])[i] = h1[i]; (&sc1[0][0])[i] = c1[i]; }
  for (int i = tid; i < KB * H2SZ; i += 256) { (&sh2[0][0])[i] = h2[i]; (&sc2[0][0])[i] = c2[i]; }
  __syncthreads();

  // LSTM1 (gates i,f,g,o); 1280 tasks
  float nh[5], nc[5];
  {
    int nt = 0;
    for (int task = tid; task < KB * H1SZ; task += 256, ++nt) {
      int b = task / H1SZ, j = task % H1SZ;
      float gi = bih1[j]            + bhh1[j];
      float gf = bih1[H1SZ + j]     + bhh1[H1SZ + j];
      float gg = bih1[2 * H1SZ + j] + bhh1[2 * H1SZ + j];
      float go = bih1[3 * H1SZ + j] + bhh1[3 * H1SZ + j];
      const int IN = ESZ + QSZ;
      for (int k = 0; k < IN; ++k) {
        float xv = sx[b][k];
        gi += Wih1[(size_t)j * IN + k] * xv;
        gf += Wih1[(size_t)(H1SZ + j) * IN + k] * xv;
        gg += Wih1[(size_t)(2 * H1SZ + j) * IN + k] * xv;
        go += Wih1[(size_t)(3 * H1SZ + j) * IN + k] * xv;
      }
      for (int k = 0; k < H1SZ; ++k) {
        float hv = sh1[b][k];
        gi += Whh1[(size_t)j * H1SZ + k] * hv;
        gf += Whh1[(size_t)(H1SZ + j) * H1SZ + k] * hv;
        gg += Whh1[(size_t)(2 * H1SZ + j) * H1SZ + k] * hv;
        go += Whh1[(size_t)(3 * H1SZ + j) * H1SZ + k] * hv;
      }
      float cn = sigf(gf) * sc1[b][j] + sigf(gi) * tanhf(gg);
      nc[nt] = cn;
      nh[nt] = sigf(go) * tanhf(cn);
    }
    __syncthreads();
    nt = 0;
    for (int task = tid; task < KB * H1SZ; task += 256, ++nt) {
      int b = task / H1SZ, j = task % H1SZ;
      sh1[b][j] = nh[nt]; sc1[b][j] = nc[nt];
    }
    __syncthreads();
  }

  // LSTM2 ; 640 tasks
  {
    int nt = 0;
    for (int task = tid; task < KB * H2SZ; task += 256, ++nt) {
      int b = task / H2SZ, j = task % H2SZ;
      float gi = bih2[j]            + bhh2[j];
      float gf = bih2[H2SZ + j]     + bhh2[H2SZ + j];
      float gg = bih2[2 * H2SZ + j] + bhh2[2 * H2SZ + j];
      float go = bih2[3 * H2SZ + j] + bhh2[3 * H2SZ + j];
      for (int k = 0; k < H1SZ; ++k) {
        float xv = sh1[b][k];
        gi += Wih2[(size_t)j * H1SZ + k] * xv;
        gf += Wih2[(size_t)(H2SZ + j) * H1SZ + k] * xv;
        gg += Wih2[(size_t)(2 * H2SZ + j) * H1SZ + k] * xv;
        go += Wih2[(size_t)(3 * H2SZ + j) * H1SZ + k] * xv;
      }
      for (int k = 0; k < H2SZ; ++k) {
        float hv = sh2[b][k];
        gi += Whh2[(size_t)j * H2SZ + k] * hv;
        gf += Whh2[(size_t)(H2SZ + j) * H2SZ + k] * hv;
        gg += Whh2[(size_t)(2 * H2SZ + j) * H2SZ + k] * hv;
        go += Whh2[(size_t)(3 * H2SZ + j) * H2SZ + k] * hv;
      }
      float cn = sigf(gf) * sc2[b][j] + sigf(gi) * tanhf(gg);
      nc[nt] = cn;
      nh[nt] = sigf(go) * tanhf(cn);
    }
    __syncthreads();
    nt = 0;
    for (int task = tid; task < KB * H2SZ; task += 256, ++nt) {
      int b = task / H2SZ, j = task % H2SZ;
      sh2[b][j] = nh[nt]; sc2[b][j] = nc[nt];
    }
    __syncthreads();
  }

  // q = h2 @ Wq^T + bq
  for (int task = tid; task < KB * QSZ; task += 256) {
    int b = task / QSZ, j = task % QSZ;
    float a = bq[j];
    for (int k = 0; k < H2SZ; ++k) a += Wq[(size_t)j * H2SZ + k] * sh2[b][k];
    sq[b][j] = a;
  }
  __syncthreads();

  // attention + context, per beam
  for (int b = 0; b < KB; ++b) {
    for (int t = tid; t < TENC; t += 256) {
      const float* kr = enc_key + (size_t)t * QSZ;
      float a = 0.0f;
      for (int q = 0; q < QSZ; ++q) a += sq[b][q] * kr[q];
      se[t] = a;
    }
    float m = -3.0e38f;
    for (int t = tid; t < TENC; t += 256) m = fmaxf(m, se[t]);
    sred[tid] = m; __syncthreads();
    for (int s = 128; s > 0; s >>= 1) {
      if (tid < s) sred[tid] = fmaxf(sred[tid], sred[tid + s]);
      __syncthreads();
    }
    float mx = sred[0]; __syncthreads();
    float sm = 0.0f;
    for (int t = tid; t < TENC; t += 256) {
      float ev = mask[t] * __expf(se[t] - mx);
      se[t] = ev; sm += ev;
    }
    sred[tid] = sm; __syncthreads();
    for (int s = 128; s > 0; s >>= 1) {
      if (tid < s) sred[tid] += sred[tid + s];
      __syncthreads();
    }
    float ssum = fmaxf(sred[0], 2e-30f); __syncthreads();
    int qi = tid & 63, grp = tid >> 6;
    float acc = 0.0f;
    for (int t = grp; t < TENC; t += 4) acc += se[t] * enc_val[(size_t)t * QSZ + qi];
    spart[grp][qi] = acc; __syncthreads();
    if (tid < QSZ)
      sctx[b][tid] = (spart[0][tid] + spart[1][tid] + spart[2][tid] + spart[3][tid]) / ssum;
    __syncthreads();
  }

  // persist state, emit A-tile x16 = [h2 | ctx] padded to 16 rows (f16)
  for (int i = tid; i < KB * H1SZ; i += 256) { h1[i] = (&sh1[0][0])[i]; c1[i] = (&sc1[0][0])[i]; }
  for (int i = tid; i < KB * H2SZ; i += 256) {
    h2[i] = (&sh2[0][0])[i]; c2[i] = (&sc2[0][0])[i]; ctx[i] = (&sctx[0][0])[i];
  }
  for (int i = tid; i < 16 * 128; i += 256) {
    int r = i >> 7, cn = i & 127;
    float v = 0.0f;
    if (r < KB) v = (cn < H2SZ) ? sh2[r][cn] : sctx[r][cn - H2SZ];
    x16[i] = (_Float16)v;
  }
}

// ---------------------------------------------------------------------------
// logits[16][VPAD] = x16[16x128] @ wc16^T + bcp.  A cached in VGPRs, TPW
// N-tiles per wave, 4x v_wmma_f32_16x16x32_f16 per tile.  Weights and bias
// are zero-padded to VPAD rows, so there is NO divergence anywhere: EXEC is
// all-ones structurally and loads can stay in flight across WMMAs.
__global__ __launch_bounds__(32) void k_logits_wmma(
    const _Float16* __restrict__ x16, const _Float16* __restrict__ wc16,
    const float* __restrict__ bcp, float* __restrict__ logits) {
  const int lane = threadIdx.x;
  const int l  = lane & 15;
  const int hi = lane >> 4;

  // A fragments: ISA 16-bit A 16x32 layout — lanes 0-15 hold K {0..7,16..23},
  // lanes 16-31 hold K {8..15,24..31}; one fragment per K-chunk of 32.
  const _Float16* arow = x16 + l * 128;
  v16h areg[4];
#pragma unroll
  for (int kc = 0; kc < 4; ++kc) {
    const int kbase = kc * 32 + hi * 8;
    v8h alo = *(const v8h*)(arow + kbase);
    v8h ahi = *(const v8h*)(arow + kbase + 16);
    areg[kc] = __builtin_shufflevector(alo, ahi, 0,1,2,3,4,5,6,7,8,9,10,11,12,13,14,15);
  }

  const int tile0 = blockIdx.x * TPW;
#pragma unroll
  for (int ti = 0; ti < TPW; ++ti) {
    const int tile = tile0 + ti;        // uniform across the wave
    if (tile >= NTILES) break;          // uniform branch (blockIdx-derived)
    const int n = tile * 16 + l;        // output column == row of padded wc16
    const _Float16* brow = wc16 + (size_t)n * 128;

    v8f acc = {};
#pragma unroll
    for (int kc = 0; kc < 4; ++kc) {
      const int kbase = kc * 32 + hi * 8;
      v8h blo = *(const v8h*)(brow + kbase);
      v8h bhi = *(const v8h*)(brow + kbase + 16);
      v16h b = __builtin_shufflevector(blo, bhi, 0,1,2,3,4,5,6,7,8,9,10,11,12,13,14,15);
      acc = __builtin_amdgcn_wmma_f32_16x16x32_f16(false, areg[kc], false, b,
                                                   (short)0, acc, false, false);
    }
    float bias = bcp[n];
#pragma unroll
    for (int r = 0; r < 8; ++r) {
      int row = hi * 8 + r;             // C/D layout: VGPR r -> M = hi*8 + r, N = l
      logits[(size_t)row * VPAD + n] = acc[r] + bias;
    }
  }
}

// ---------------------------------------------------------------------------
// Grid-parallel partial max / sumexp per beam slice (grid: MSLICE x KB).
__global__ __launch_bounds__(256) void k_partms(const float* __restrict__ logits,
                                                float* __restrict__ partM,
                                                float* __restrict__ partS) {
  const int b = blockIdx.y, s = blockIdx.x, tid = threadIdx.x;
  const int lo = s * MCH;
  const int hi = (lo + MCH < VSZ) ? lo + MCH : VSZ;
  __shared__ float red[256];
  float m = -3.0e38f;
  for (int t = lo + tid; t < hi; t += 256) m = fmaxf(m, logits[(size_t)b * VPAD + t]);
  red[tid] = m; __syncthreads();
  for (int r = 128; r > 0; r >>= 1) {
    if (tid < r) red[tid] = fmaxf(red[tid], red[tid + r]);
    __syncthreads();
  }
  float mx = red[0]; __syncthreads();
  float sm = 0.0f;
  for (int t = lo + tid; t < hi; t += 256) sm += __expf(logits[(size_t)b * VPAD + t] - mx);
  red[tid] = sm; __syncthreads();
  for (int r = 128; r > 0; r >>= 1) {
    if (tid < r) red[tid] += red[tid + r];
    __syncthreads();
  }
  if (tid == 0) { partM[b * MSLICE + s] = mx; partS[b * MSLICE + s] = red[0]; }
}

// ---------------------------------------------------------------------------
// Per-block top-10 over a contiguous slice of the KB*V candidate space.
// Candidate values computed on the fly (no rewrite pass over logits).
__global__ __launch_bounds__(256) void k_blocktop(
    const float* __restrict__ logits,
    const float* __restrict__ partM, const float* __restrict__ partS,
    const float* __restrict__ stF, const int* __restrict__ stI,
    float* __restrict__ topv, int* __restrict__ topi, int step) {
  const int tid = threadIdx.x;
  __shared__ float slz[KB], ssc[KB];
  __shared__ int   sfin[KB], sel[KB];
  __shared__ float redv[256];
  __shared__ int   redi[256];

  if (tid < KB) {
    float M = -3.0e38f, S = 0.0f;  // online merge of MSLICE partials
    for (int s = 0; s < MSLICE; ++s) {
      float m = partM[tid * MSLICE + s], sv = partS[tid * MSLICE + s];
      if (m > M) { S = S * __expf(M - m) + sv; M = m; }
      else       { S += sv * __expf(m - M); }
    }
    slz[tid] = M + __logf(S);
    ssc[tid] = stF[4480 + tid];
    sfin[tid] = stI[KB + tid];
  }
  __syncthreads();

  const int chunk = (TOTC + NTB - 1) / NTB;
  const int lo = blockIdx.x * chunk;
  const int hi = (lo + chunk < TOTC) ? lo + chunk : TOTC;

  for (int s = 0; s < KB; ++s) {
    float bv = -3.4e38f; int bi = TOTC;
    for (int idx = lo + tid; idx < hi; idx += 256) {
      bool skip = false;
      for (int u = 0; u < s; ++u) skip = skip || (sel[u] == idx);
      if (skip) continue;
      int b = idx / VSZ, t = idx - b * VSZ;
      float v;
      if (sfin[b]) v = (t == EOSI) ? ssc[b] : NEGF;
      else         v = ssc[b] + logits[(size_t)b * VPAD + t] - slz[b];
      if (step == 0 && b > 0) v = NEGF;
      if (v > bv || (v == bv && idx < bi)) { bv = v; bi = idx; }
    }
    redv[tid] = bv; redi[tid] = bi; __syncthreads();
    for (int r = 128; r > 0; r >>= 1) {
      if (tid < r) {
        float ov = redv[tid + r]; int oi = redi[tid + r];
        if (ov > redv[tid] || (ov == redv[tid] && oi < redi[tid])) {
          redv[tid] = ov; redi[tid] = oi;
        }
      }
      __syncthreads();
    }
    if (tid == 0) {
      sel[s] = redi[0];
      topv[blockIdx.x * KB + s] = redv[0];
      topi[blockIdx.x * KB + s] = redi[0];
    }
    __syncthreads();
  }
}

// ---------------------------------------------------------------------------
// Merge NC block candidates -> global top-10, then permute/update beam state.
__global__ __launch_bounds__(256) void k_final_update(
    float* __restrict__ stF, int* __restrict__ stI,
    const float* __restrict__ topv, const int* __restrict__ topi, int step) {
  const int tid = threadIdx.x;
  float* h1 = stF;        float* c1 = stF + 1280;
  float* h2 = stF + 2560; float* c2 = stF + 3200;
  float* ctx = stF + 3840; float* scores = stF + 4480;
  int* prev = stI; int* fin = stI + KB; int* seqs = stI + 2 * KB;

  __shared__ float cv[NC];
  __shared__ int   ci[NC];
  __shared__ unsigned char used[NC];
  __shared__ float selv[KB], sscore[KB];
  __shared__ int   seli[KB], sfin[KB];
  __shared__ float redv[256];
  __shared__ int   redi[256], reds[256];
  __shared__ float sstate[KB * 448];
  __shared__ int   sseq[KB * (MAXLEN + 1)];

  for (int i = tid; i < NC; i += 256) { cv[i] = topv[i]; ci[i] = topi[i]; used[i] = 0; }
  if (tid < KB) { sscore[tid] = scores[tid]; sfin[tid] = fin[tid]; }
  __syncthreads();

  // global top-10 over NC entries (global indices unique across blocks)
  for (int s = 0; s < KB; ++s) {
    float bv = -3.5e38f; int bi = 0x7fffffff; int bslot = -1;
    for (int i = tid; i < NC; i += 256) {
      if (used[i]) continue;
      float v = cv[i]; int idx = ci[i];
      if (v > bv || (v == bv && idx < bi)) { bv = v; bi = idx; bslot = i; }
    }
    redv[tid] = bv; redi[tid] = bi; reds[tid] = bslot; __syncthreads();
    for (int r = 128; r > 0; r >>= 1) {
      if (tid < r) {
        float ov = redv[tid + r]; int oi = redi[tid + r];
        if (ov > redv[tid] || (ov == redv[tid] && oi < redi[tid])) {
          redv[tid] = ov; redi[tid] = oi; reds[tid] = reds[tid + r];
        }
      }
      __syncthreads();
    }
    if (tid == 0) {
      selv[s] = redv[0]; seli[s] = redi[0];
      if (reds[0] >= 0) used[reds[0]] = 1;
    }
    __syncthreads();
  }

  // snapshot old state
  for (int i = tid; i < KB * H1SZ; i += 256) {
    int b = i / H1SZ, j = i % H1SZ;
    sstate[b * 448 + j] = h1[i];
    sstate[b * 448 + 128 + j] = c1[i];
  }
  for (int i = tid; i < KB * H2SZ; i += 256) {
    int b = i / H2SZ, j = i % H2SZ;
    sstate[b * 448 + 256 + j] = h2[i];
    sstate[b * 448 + 320 + j] = c2[i];
    sstate[b * 448 + 384 + j] = ctx[i];
  }
  for (int i = tid; i < KB * (MAXLEN + 1); i += 256) sseq[i] = seqs[i];
  __syncthreads();

  // permute/gather into new state
  for (int i = tid; i < KB * 448; i += 256) {
    int s = i / 448, j = i - s * 448;
    int pb = seli[s] / VSZ; pb = (pb < KB) ? pb : KB - 1;
    float v = sstate[pb * 448 + j];
    if (j < 128)      h1[s * 128 + j] = v;
    else if (j < 256) c1[s * 128 + (j - 128)] = v;
    else if (j < 320) h2[s * 64 + (j - 256)] = v;
    else if (j < 384) c2[s * 64 + (j - 320)] = v;
    else              ctx[s * 64 + (j - 384)] = v;
  }
  for (int i = tid; i < KB * (MAXLEN + 1); i += 256) {
    int s = i / (MAXLEN + 1), j = i - s * (MAXLEN + 1);
    int pb = seli[s] / VSZ; pb = (pb < KB) ? pb : KB - 1;
    seqs[i] = sseq[pb * (MAXLEN + 1) + j];
  }
  __syncthreads();
  if (tid < KB) {
    int pb = seli[tid] / VSZ; pb = (pb < KB) ? pb : KB - 1;
    int nxt = seli[tid] - (seli[tid] / VSZ) * VSZ;
    seqs[tid * (MAXLEN + 1) + step + 1] = nxt;
    prev[tid] = nxt;
    scores[tid] = selv[tid];
    fin[tid] = (sfin[pb] || nxt == EOSI) ? 1 : 0;
  }
}

// ---------------------------------------------------------------------------
__global__ __launch_bounds__(320) void k_finalize(const float* __restrict__ stF,
                                                  const int* __restrict__ stI,
                                                  float* __restrict__ out) {
  int tid = threadIdx.x;
  const float* scores = stF + 4480;
  const int* seqs = stI + 2 * KB;
  if (tid < KB * (MAXLEN + 1)) out[tid] = (float)seqs[tid];
  if (tid < KB) {
    out[KB * (MAXLEN + 1) + tid] = scores[tid];
    int first = -1;
    for (int j = 0; j < MAXLEN; ++j)
      if (seqs[tid * (MAXLEN + 1) + 1 + j] == EOSI) { first = j; break; }
    float len = (first >= 0) ? (float)(first + 2) : (float)(MAXLEN + 2);
    out[KB * (MAXLEN + 1) + KB + tid] = scores[tid] / powf(len, 1.2f);
  }
}

// ---------------------------------------------------------------------------
extern "C" void kernel_launch(void* const* d_in, const int* in_sizes, int n_in,
                              void* d_out, int out_size, void* d_ws, size_t ws_size,
                              hipStream_t stream) {
  const float* enc_key = (const float*)d_in[0];
  const float* enc_val = (const float*)d_in[1];
  const float* mask    = (const float*)d_in[2];
  const float* emb     = (const float*)d_in[3];
  const float* Wih1    = (const float*)d_in[4];
  const float* Whh1    = (const float*)d_in[5];
  const float* bih1    = (const float*)d_in[6];
  const float* bhh1    = (const float*)d_in[7];
  const float* Wih2    = (const float*)d_in[8];
  const float* Whh2    = (const float*)d_in[9];
  const float* bih2    = (const float*)d_in[10];
  const float* bhh2    = (const float*)d_in[11];
  const float* Wq      = (const float*)d_in[12];
  const float* bq      = (const float*)d_in[13];
  const float* Wc      = (const float*)d_in[14];
  const float* bc      = (const float*)d_in[15];

  char* ws = (char*)d_ws;
  _Float16* wc16  = (_Float16*)ws;
  _Float16* x16   = (_Float16*)(ws + OFF_X16);
  float*    logit = (float*)(ws + OFF_LOG);
  float*    stF   = (float*)(ws + OFF_STF);
  int*      stI   = (int*)(ws + OFF_STI);
  float*    partM = (float*)(ws + OFF_PM);
  float*    partS = (float*)(ws + OFF_PS);
  float*    topv  = (float*)(ws + OFF_TV);
  int*      topi  = (int*)(ws + OFF_TI);
  float*    bcp   = (float*)(ws + OFF_BC);

  k_cvt_wc<<<(VPAD * 128 + 255) / 256, 256, 0, stream>>>(Wc, bc, wc16, bcp);
  k_init_state<<<1, 256, 0, stream>>>(stF, stI);

  for (int step = 0; step < MAXLEN; ++step) {
    k_step_rnn<<<1, 256, 0, stream>>>(emb, Wih1, Whh1, bih1, bhh1,
                                      Wih2, Whh2, bih2, bhh2,
                                      Wq, bq, enc_key, enc_val, mask,
                                      stF, stI, x16);
    k_logits_wmma<<<NWBLK, 32, 0, stream>>>(x16, wc16, bcp, logit);
    k_partms<<<dim3(MSLICE, KB), 256, 0, stream>>>(logit, partM, partS);
    k_blocktop<<<NTB, 256, 0, stream>>>(logit, partM, partS, stF, stI, topv, topi, step);
    k_final_update<<<1, 256, 0, stream>>>(stF, stI, topv, topi, step);
  }
  k_finalize<<<1, 320, 0, stream>>>(stF, stI, (float*)d_out);
}
